// RecurrentFixed_PredPrey_38079180046777
// MI455X (gfx1250) — matrix operand order repaired and checked by reading, not compiled, not run on hardware
//
#include <hip/hip_runtime.h>

typedef __attribute__((ext_vector_type(16))) _Float16 v16h;
typedef __attribute__((ext_vector_type(8)))  float    v8f;
typedef __attribute__((ext_vector_type(8)))  unsigned v8u;
typedef __attribute__((ext_vector_type(2)))  __fp16   v2hp;

#define HDIM   256
#define NTILE  16      // 16 feature tiles of 16
#define KCH    8       // 8 K-chunks of 32
#define NLAYER 30
#define WAVES  8       // waves per block
#define TPB    (WAVES * 32)
#define BROWS  32768

// LDS: swizzled A operand (Ww) in exact WMMA A-layout, 128 tiles x 32 lanes x 16 halves
#define A_LDS_HALVES (NTILE * KCH * 512)   // 65536 halves = 128 KB

__device__ __forceinline__ float fix_pred(float p) { return (p == -1.0f) ? 0.0f : p; }

// tanh(s*v) = 1 - 2/(exp2(v * 2*s*log2e) + 1)   (saturating, NaN-free for +-inf exp)
__device__ __forceinline__ float tanh_fast(float v, float c) {
  float e = __builtin_amdgcn_exp2f(v * c);
  return 1.0f - 2.0f * __builtin_amdgcn_rcpf(e + 1.0f);
}

__device__ __forceinline__ unsigned pack_h2(float a, float b) {
  v2hp t = __builtin_amdgcn_cvt_pkrtz(a, b);
  return __builtin_bit_cast(unsigned, t);
}

extern "C" __global__ void __launch_bounds__(TPB)
__attribute__((amdgpu_waves_per_eu(3)))
pp_kernel(const float* __restrict__ X, const float* __restrict__ pred,
          const float* __restrict__ W, const float* __restrict__ w,
          const float* __restrict__ a, const float* __restrict__ bias,
          const float* __restrict__ scalar, float* __restrict__ out)
{
  extern __shared__ __align__(16) _Float16 sA[];

  const int tid  = threadIdx.x;
  const int lane = tid & 31;
  const int wave = tid >> 5;
  const int h    = lane >> 4;   // lane half
  const int n    = lane & 15;   // batch column within wave's 16-row block
  const size_t nrow = (size_t)(blockIdx.x * WAVES + wave) * 16 + n;

  const float tc = scalar[0] * 2.0f * 1.44269504088896f; // 2*s*log2(e)

  // ---- Build swizzled A = Ww = w[:,None]*W_hidden into LDS (A-operand layout) ----
  // tile (t,c): A[i,k] = Ww[16t+i, 32c+k]; lane L stores 16 halves:
  // vgpr v pair k0 = 2v + (L>=16?16:0)   (contiguous k, matches B layout)
  for (int f = tid; f < A_LDS_HALVES; f += TPB) {
    const int tile   = f >> 9;          // 0..127
    const int within = f & 511;
    const int L  = within >> 4;
    const int e  = within & 15;
    const int v  = e >> 1;
    const int hh = e & 1;
    const int t  = tile >> 3;
    const int c  = tile & 7;
    const int i  = L & 15;
    const int k0 = 2 * v + ((L >= 16) ? 16 : 0);
    const int row = 16 * t + i;
    const int col = 32 * c + k0 + hh;
    sA[f] = (_Float16)(w[row] * W[row * HDIM + col]);
  }
  __syncthreads();

  // ---- Register-resident drive^T: drv[t][r] = element (m = 16t + 8h + r, nrow) ----
  float drv[NTILE][8];
  #pragma unroll
  for (int t = 0; t < NTILE; ++t) {
    const int m0 = t * 16 + 8 * h;
    const float4 x0 = *(const float4*)(X    + nrow * HDIM + m0);
    const float4 x1 = *(const float4*)(X    + nrow * HDIM + m0 + 4);
    const float4 p0 = *(const float4*)(pred + nrow * HDIM + m0);
    const float4 p1 = *(const float4*)(pred + nrow * HDIM + m0 + 4);
    const float4 a0 = *(const float4*)(a    + m0);
    const float4 a1 = *(const float4*)(a    + m0 + 4);
    const float4 b0 = *(const float4*)(bias + m0);
    const float4 b1 = *(const float4*)(bias + m0 + 4);
    drv[t][0] = fix_pred(p0.x) + b0.x + a0.x * x0.x;
    drv[t][1] = fix_pred(p0.y) + b0.y + a0.y * x0.y;
    drv[t][2] = fix_pred(p0.z) + b0.z + a0.z * x0.z;
    drv[t][3] = fix_pred(p0.w) + b0.w + a0.w * x0.w;
    drv[t][4] = fix_pred(p1.x) + b1.x + a1.x * x1.x;
    drv[t][5] = fix_pred(p1.y) + b1.y + a1.y * x1.y;
    drv[t][6] = fix_pred(p1.z) + b1.z + a1.z * x1.z;
    drv[t][7] = fix_pred(p1.w) + b1.w + a1.w * x1.w;
  }

  // ---- B operand init: u0^T = pred^T in f16 B-layout.
  // lane half h, vgpr v of chunk c holds K = 32c + 16h + 2v, +1 (batch col = n)
  v8u Ba[KCH];
  v8u Bb[KCH] = {};
  #pragma unroll
  for (int c = 0; c < KCH; ++c) {
    #pragma unroll
    for (int v = 0; v < 8; ++v) {
      const float2 pp = *(const float2*)(pred + nrow * HDIM + c * 32 + h * 16 + 2 * v);
      Ba[c][v] = pack_h2(pp.x, pp.y);
    }
  }

  // ---- One iteration: D_t = drive_t + sum_c A(t,c) x B(c); u = tanh; repack B ----
  auto step = [&](const v8u (&Bi)[KCH], v8u (&Bo)[KCH]) {
    // Opaque zero: stops LICM/CSE from hoisting the 128 LDS tile loads out of
    // the layer loop and caching them in (spilled) registers.
    int zoff = 0;
    asm volatile("" : "+v"(zoff));
    const _Float16* sAz = sA + zoff;
    #pragma unroll
    for (int t = 0; t < NTILE; ++t) {
      // Stage all 8 A tiles first -> one load clause, partial dscnt waits,
      // chunk-c load latency hides behind WMMAs 0..c-1.
      v16h At[KCH];
      #pragma unroll
      for (int c = 0; c < KCH; ++c)
        At[c] = *(const v16h*)(sAz + ((t * KCH + c) << 9) + (lane << 4));
      v8f acc;
      #pragma unroll
      for (int r = 0; r < 8; ++r) acc[r] = drv[t][r];
      #pragma unroll
      for (int c = 0; c < KCH; ++c) {
        const v16h Bc = __builtin_bit_cast(v16h, Bi[c]);
        acc = __builtin_amdgcn_wmma_f32_16x16x32_f16(false, At[c], false, Bc,
                                                     (short)0, acc, false, false);
      }
      // tile t feeds chunk cc = t>>1, lane-half s = t&1 of the next B operand.
      // source half h contributes vgprs 4h..4h+3; other 4 come via half-swap.
      const int  s   = t & 1;
      const int  cc  = t >> 1;
      const bool upd = (h == s);
      #pragma unroll
      for (int j = 0; j < 4; ++j) {
        const float    u0   = tanh_fast(acc[2 * j],     tc);
        const float    u1   = tanh_fast(acc[2 * j + 1], tc);
        const unsigned mine = pack_h2(u0, u1);
        // ds_swizzle SWAPX16: group-of-32 xor_mask=0x10, and_mask=0x1f
        const unsigned othr =
            (unsigned)__builtin_amdgcn_ds_swizzle((int)mine, 0x401f);
        const unsigned lo   = h ? othr : mine;   // slot j   (source half 0)
        const unsigned hi   = h ? mine : othr;   // slot j+4 (source half 1)
        Bo[cc][j]     = upd ? lo : Bo[cc][j];
        Bo[cc][j + 4] = upd ? hi : Bo[cc][j + 4];
      }
    }
  };

  // ---- Final iteration: write f32 u straight to global (two float4 per tile) ----
  auto final_step = [&](const v8u (&Bi)[KCH]) {
    int zoff = 0;
    asm volatile("" : "+v"(zoff));
    const _Float16* sAz = sA + zoff;
    #pragma unroll
    for (int t = 0; t < NTILE; ++t) {
      v16h At[KCH];
      #pragma unroll
      for (int c = 0; c < KCH; ++c)
        At[c] = *(const v16h*)(sAz + ((t * KCH + c) << 9) + (lane << 4));
      v8f acc;
      #pragma unroll
      for (int r = 0; r < 8; ++r) acc[r] = drv[t][r];
      #pragma unroll
      for (int c = 0; c < KCH; ++c) {
        const v16h Bc = __builtin_bit_cast(v16h, Bi[c]);
        acc = __builtin_amdgcn_wmma_f32_16x16x32_f16(false, At[c], false, Bc,
                                                     (short)0, acc, false, false);
      }
      float4 o0, o1;
      o0.x = tanh_fast(acc[0], tc); o0.y = tanh_fast(acc[1], tc);
      o0.z = tanh_fast(acc[2], tc); o0.w = tanh_fast(acc[3], tc);
      o1.x = tanh_fast(acc[4], tc); o1.y = tanh_fast(acc[5], tc);
      o1.z = tanh_fast(acc[6], tc); o1.w = tanh_fast(acc[7], tc);
      *(float4*)(out + nrow * HDIM + t * 16 + 8 * h)     = o0;
      *(float4*)(out + nrow * HDIM + t * 16 + 8 * h + 4) = o1;
    }
  };

  #pragma unroll 1
  for (int l = 0; l < 14; ++l) {   // 28 layers, ping-pong
    step(Ba, Bb);
    step(Bb, Ba);
  }
  step(Ba, Bb);                    // layer 29
  final_step(Bb);                  // layer 30 -> global
}

extern "C" void kernel_launch(void* const* d_in, const int* in_sizes, int n_in,
                              void* d_out, int out_size, void* d_ws, size_t ws_size,
                              hipStream_t stream) {
  const float* X      = (const float*)d_in[0];
  const float* pred   = (const float*)d_in[1];
  const float* W      = (const float*)d_in[2];
  const float* w      = (const float*)d_in[3];
  const float* a      = (const float*)d_in[4];
  const float* bias   = (const float*)d_in[5];
  const float* scalar = (const float*)d_in[6];
  float* out = (float*)d_out;

  const size_t shmem = (size_t)A_LDS_HALVES * sizeof(_Float16); // 128 KB
  (void)hipFuncSetAttribute((const void*)pp_kernel,
                            hipFuncAttributeMaxDynamicSharedMemorySize, (int)shmem);

  dim3 grid(BROWS / (16 * WAVES));  // 256 blocks x 8 waves = 2048 waves (16 rows each)
  pp_kernel<<<grid, TPB, shmem, stream>>>(X, pred, W, w, a, bias, scalar, out);
}
